// GATConv_86131274154633
// MI455X (gfx1250) — compile-verified, matrix operand order checked by reading, not run on hardware
//
#include <hip/hip_runtime.h>

#define NNODES 50000
#define ERAND  800000
#define ETOT   (ERAND + NNODES)   // 850000 (self-loops appended)
#define INSZ   512
#define OUTC   64
#define HEADS  8
#define CHN    (OUTC * HEADS)     // 512
#define NEG_SLOPE 0.01f

typedef __attribute__((ext_vector_type(16))) _Float16 v16h;
typedef __attribute__((ext_vector_type(8)))  float    v8f;

// ---------------- fp32 -> fp16 cast (feeds WMMA) ----------------
__global__ void cast_f32_to_f16(const float* __restrict__ src,
                                _Float16* __restrict__ dst, int n) {
  int i = blockIdx.x * blockDim.x + threadIdx.x;
  if (i < n) dst[i] = (_Float16)src[i];
}

// ---------------- Z = X @ W^T + b via v_wmma_f32_16x16x32_f16 ----------------
// One wave computes one 16(M=nodes) x 16(N=channels) tile, K-loop over 512 in
// steps of 32. A: lane L holds X[m0+(L&15)][k0+(L>>4)*16 .. +16] (contiguous).
// B = W^T, so B[k][n] = W[n0+n][k]: lane L holds W[n0+(L&15)][k0+(L>>4)*16 .. +16]
// (also contiguous) — A and B use the same contiguous k order, so the k products pair correctly.
__global__ __launch_bounds__(256)
void gemm_wmma_f16(const _Float16* __restrict__ Xh, const _Float16* __restrict__ Wh,
                   const float* __restrict__ bias, float* __restrict__ Z) {
  const int lane = threadIdx.x & 31;
  const int wave = threadIdx.x >> 5;
  const int m0 = blockIdx.x << 4;                 // node tile (3125 tiles, exact)
  const int n0 = (blockIdx.y * 8 + wave) << 4;    // channel tile (32 tiles)
  const int sub = lane >> 4;                      // 0: K low half, 1: K high half
  const int l15 = lane & 15;

  const _Float16* ap = Xh + (size_t)(m0 + l15) * INSZ + sub * 16;
  const _Float16* bp = Wh + (size_t)(n0 + l15) * INSZ + sub * 16;

  v8f acc = {};
#pragma unroll 4
  for (int k0 = 0; k0 < INSZ; k0 += 32) {
    v16h a = *(const v16h*)(ap + k0);
    v16h b = *(const v16h*)(bp + k0);
    acc = __builtin_amdgcn_wmma_f32_16x16x32_f16(false, a, false, b,
                                                 (short)0, acc, false, false);
  }

  // C/D layout: VGPR r -> M = r + 8*(lane>>4), N = lane&15
  const int colc = n0 + l15;
  const float bv = bias[colc];
  float* zp = Z + (size_t)(m0 + sub * 8) * CHN + colc;
#pragma unroll
  for (int r = 0; r < 8; ++r)
    zp[(size_t)r * CHN] = acc[r] + bv;
}

// ---------------- e_l[n,h] = sum_o Z[n,o,h]*a_l[o,h]; same for e_r ----------------
__global__ void attn_logits(const float* __restrict__ Z, const float* __restrict__ a_l,
                            const float* __restrict__ a_r, float* __restrict__ el,
                            float* __restrict__ er) {
  int t = blockIdx.x * blockDim.x + threadIdx.x;
  if (t >= NNODES * HEADS) return;
  int n = t >> 3, h = t & 7;
  const float* zrow = Z + (size_t)n * CHN + h;
  float sl = 0.f, sr = 0.f;
#pragma unroll
  for (int o = 0; o < OUTC; ++o) {
    float z = zrow[o * HEADS];
    sl += z * a_l[o * HEADS + h];
    sr += z * a_r[o * HEADS + h];
  }
  el[t] = sl;
  er[t] = sr;
}

// ---------------- helpers ----------------
__device__ __forceinline__ float lrelu(float v) { return v > 0.f ? v : NEG_SLOPE * v; }

// monotone float <-> uint key so atomicMax(u32) implements float max; key 0 == -inf
__device__ __forceinline__ unsigned fkey(float f) {
  unsigned u = __float_as_uint(f);
  return (u & 0x80000000u) ? ~u : (u | 0x80000000u);
}
__device__ __forceinline__ float funkey(unsigned k) {
  return __uint_as_float((k & 0x80000000u) ? (k & 0x7FFFFFFFu) : ~k);
}

// ---------------- pass 1: segment max over edges grouped by row ----------------
__global__ void edge_max(const int* __restrict__ row, const int* __restrict__ col,
                         const float* __restrict__ el, const float* __restrict__ er,
                         unsigned* __restrict__ mkey) {
  int t = blockIdx.x * blockDim.x + threadIdx.x;
  if (t >= ETOT * HEADS) return;
  int e = t >> 3, h = t & 7;
  int r = row[e], c = col[e];
  float v = lrelu(el[r * HEADS + h] + er[c * HEADS + h]);
  atomicMax(&mkey[r * HEADS + h], fkey(v));
}

// ---------------- pass 2: denom[row,h] += exp(e - m) ----------------
__global__ void edge_denom(const int* __restrict__ row, const int* __restrict__ col,
                           const float* __restrict__ el, const float* __restrict__ er,
                           const unsigned* __restrict__ mkey, float* __restrict__ denom) {
  int t = blockIdx.x * blockDim.x + threadIdx.x;
  if (t >= ETOT * HEADS) return;
  int e = t >> 3, h = t & 7;
  int r = row[e], c = col[e];
  float v = lrelu(el[r * HEADS + h] + er[c * HEADS + h]);
  float ex = __expf(v - funkey(mkey[r * HEADS + h]));
  atomicAdd(&denom[r * HEADS + h], ex);
}

// ---------------- pass 3: out[row] += coef * Z[col]  (one wave per edge) ----------
// Z rows (2KB each, 102MB total) are L2-resident on MI455X (192MB L2), so the
// 1.74GB of random row gathers here is served from L2, not HBM.
__global__ __launch_bounds__(256)
void edge_scatter(const int* __restrict__ row, const int* __restrict__ col,
                  const float* __restrict__ el, const float* __restrict__ er,
                  const unsigned* __restrict__ mkey, const float* __restrict__ denom,
                  const float* __restrict__ Z, float* __restrict__ out) {
  int e = blockIdx.x * 8 + (threadIdx.x >> 5);
  if (e >= ETOT) return;
  int lane = threadIdx.x & 31;
  int r = row[e], c = col[e];
  int h = lane & 7;  // lanes 0..7 compute the 8 head coefficients (replicated x4)
  float v = lrelu(el[r * HEADS + h] + er[c * HEADS + h]);
  float coef = __expf(v - funkey(mkey[r * HEADS + h])) / denom[r * HEADS + h];
  const float* zrow = Z + (size_t)c * CHN;
  float* orow = out + (size_t)r * CHN;
  int base = lane * 16;  // each lane covers 16 contiguous channels; ch&7 == j&7
#pragma unroll
  for (int j = 0; j < 16; ++j) {
    float w = __shfl(coef, j & 7, 8);
    int ch = base + j;
    atomicAdd(&orow[ch], w * zrow[ch]);
  }
}

extern "C" void kernel_launch(void* const* d_in, const int* in_sizes, int n_in,
                              void* d_out, int out_size, void* d_ws, size_t ws_size,
                              hipStream_t stream) {
  const float* X  = (const float*)d_in[0];
  const float* W  = (const float*)d_in[1];
  const float* b  = (const float*)d_in[2];
  const float* al = (const float*)d_in[3];
  const float* ar = (const float*)d_in[4];
  const int* row  = (const int*)d_in[5];
  const int* col  = (const int*)d_in[6];
  (void)in_sizes; (void)n_in; (void)ws_size;

  // workspace carve-out (~160 MB total)
  char* ws = (char*)d_ws;
  size_t off = 0;
  auto take = [&](size_t bytes) -> char* {
    char* p = ws + off;
    off = (off + bytes + 255) & ~(size_t)255;
    return p;
  };
  _Float16* Xh  = (_Float16*)take((size_t)NNODES * INSZ * 2);  // 51.2 MB
  _Float16* Wh  = (_Float16*)take((size_t)CHN * INSZ * 2);     // 0.5 MB
  float*    Z   = (float*)   take((size_t)NNODES * CHN * 4);   // 102.4 MB (L2-resident)
  float*    el  = (float*)   take((size_t)NNODES * HEADS * 4);
  float*    er  = (float*)   take((size_t)NNODES * HEADS * 4);
  unsigned* mk  = (unsigned*)take((size_t)NNODES * HEADS * 4);
  float*    den = (float*)   take((size_t)NNODES * HEADS * 4);

  // zero the accumulators (mk=0 encodes -inf under the key mapping)
  hipMemsetAsync(d_out, 0, (size_t)out_size * sizeof(float), stream);
  hipMemsetAsync(mk,  0, (size_t)NNODES * HEADS * 4, stream);
  hipMemsetAsync(den, 0, (size_t)NNODES * HEADS * 4, stream);

  const int nX = NNODES * INSZ;
  cast_f32_to_f16<<<(nX + 255) / 256, 256, 0, stream>>>(X, Xh, nX);
  const int nW = CHN * INSZ;
  cast_f32_to_f16<<<(nW + 255) / 256, 256, 0, stream>>>(W, Wh, nW);

  dim3 gg(NNODES / 16, CHN / (16 * 8));          // (3125, 4), 8 waves/block
  gemm_wmma_f16<<<gg, 256, 0, stream>>>(Xh, Wh, b, Z);

  attn_logits<<<(NNODES * HEADS + 255) / 256, 256, 0, stream>>>(Z, al, ar, el, er);

  const int eh = ETOT * HEADS;                    // 6.8M threads
  edge_max<<<(eh + 255) / 256, 256, 0, stream>>>(row, col, el, er, mk);
  edge_denom<<<(eh + 255) / 256, 256, 0, stream>>>(row, col, el, er, mk, den);
  edge_scatter<<<(ETOT + 7) / 8, 256, 0, stream>>>(row, col, el, er, mk, den, Z,
                                                   (float*)d_out);
}
